// Attention_21345987461594
// MI455X (gfx1250) — matrix operand order repaired and checked by reading, not compiled, not run on hardware
//
#include <hip/hip_runtime.h>
#include <hip/hip_bf16.h>

// B=2, S=4096, E=512, H=8, D=64
#define BB 2
#define SS 4096
#define EE 512
#define HH 8
#define DD 64

typedef __attribute__((ext_vector_type(16))) __bf16 v16bf;
typedef __attribute__((ext_vector_type(8)))  float  v8f;
typedef __attribute__((ext_vector_type(4)))  unsigned v4u;
typedef __attribute__((ext_vector_type(8)))  int      v8i;
typedef __attribute__((ext_vector_type(4)))  int      v4i;

// ---------------------------------------------------------------------------
// Tensor Data Mover (gfx1250). Guarded so the file compiles on toolchains
// without the builtin; arity differs between ROCm 7.2 (5 args) and
// amdgpu-toolchain (6 args) -- discriminate via the TDM header that only
// ships with the latter.
// ---------------------------------------------------------------------------
#if __has_builtin(__builtin_amdgcn_tensor_load_to_lds)
#define HAS_TDM 1
#if __has_include(<hip/amd_detail/amd_gfx1250_TDM.h>)
#define TDM_CALL(g0, g1, a, b, z8, pol) __builtin_amdgcn_tensor_load_to_lds(g0, g1, a, b, z8, pol)
#else
#define TDM_CALL(g0, g1, a, b, z8, pol) __builtin_amdgcn_tensor_load_to_lds(g0, g1, a, b, pol)
#endif
#else
#define HAS_TDM 0
#endif

#if __has_builtin(__builtin_amdgcn_s_wait_tensorcnt)
#define WAIT_TENSOR() __builtin_amdgcn_s_wait_tensorcnt(0)
#else
#define WAIT_TENSOR() asm volatile("s_wait_tensorcnt 0" ::: "memory")
#endif

#if HAS_TDM
// Issue a 2D bf16 tile DMA: global [dim1 x dim0] row-major (stride0 elems
// between rows) -> LDS, tile [tile1 x tile0], tile start at gptr.
// D# packing per CDNA5 ISA Sec 8.3/8.4.
__device__ __forceinline__ void tdm_load_2d(unsigned lds_off, const void* gptr,
                                            unsigned dim0, unsigned dim1,
                                            unsigned tile0, unsigned tile1,
                                            unsigned stride0) {
  unsigned long long ga = (unsigned long long)(uintptr_t)gptr;
  v4u g0 = { 1u,                                   // count=1, user mode
             lds_off,                              // lds_addr (bytes)
             (unsigned)ga,                         // global_addr[31:0]
             ((unsigned)(ga >> 32) & 0x01FFFFFFu) | 0x80000000u };  // addr[56:32] | type=2
  v8i g1 = { (int)(1u << 16),                      // data_size=1 -> 2 bytes
             (int)((dim0 & 0xFFFFu) << 16),        // tensor_dim0[15:0]
             (int)((dim0 >> 16) | ((dim1 & 0xFFFFu) << 16)),   // dim0[31:16] | dim1[15:0]
             (int)((dim1 >> 16) | ((tile0 & 0xFFFFu) << 16)),  // dim1[31:16] | tile_dim0
             (int)(tile1 & 0xFFFFu),               // tile_dim1 (tile_dim2=0)
             (int)stride0,                         // tensor_dim0_stride[31:0]
             0, 0 };
  v4i z4 = { 0, 0, 0, 0 };
  v8i z8 = { 0, 0, 0, 0, 0, 0, 0, 0 };
  TDM_CALL(g0, g1, z4, z4, z8, 0);
}
#endif

__device__ __forceinline__ unsigned lds_off_of(const void* p) {
  return (unsigned)(uintptr_t)p;  // low 32 bits of shared aperture addr = LDS offset
}

__device__ __forceinline__ __bf16 f2bf(float f) {
  union { float f; unsigned u; } v; v.f = f;
  unsigned r = (v.u + 0x7FFFu + ((v.u >> 16) & 1u)) >> 16;
  union { unsigned short s; __bf16 b; } o; o.s = (unsigned short)r;
  return o.b;
}

// A-fragment 16x32 bf16 (M x K), source LDS tile row-major [row][k], stride lda.
// element e<8 -> K = e + hi*8 ; e>=8 -> K = e + 8 + hi*8   (ISA 7.12.2)
__device__ __forceinline__ v16bf load_frag_a(const __bf16* __restrict__ a, int lda, int lane) {
  int m  = lane & 15;
  int hi = (lane >> 4) & 1;
  const __bf16* row = a + m * lda;
  v16bf f;
#pragma unroll
  for (int e = 0; e < 8; ++e)  f[e] = row[e + hi * 8];
#pragma unroll
  for (int e = 8; e < 16; ++e) f[e] = row[e + 8 + hi * 8];
  return f;
}

// B-fragment 32x16 bf16 (K x N), source LDS tile TRANSPOSED row-major [n][k].
// lane n = lane&15 ; e<8 -> K = e + hi*8 ; e>=8 -> K = e + 8 + hi*8
__device__ __forceinline__ v16bf load_frag_b(const __bf16* __restrict__ bt, int ldb, int lane) {
  int n  = lane & 15;
  int hi = (lane >> 4) & 1;
  const __bf16* row = bt + n * ldb + hi * 8;
  v16bf f;
#pragma unroll
  for (int e = 0; e < 16; ++e) f[e] = row[e < 8 ? e : e + 8];
  return f;
}

#define WMMA_BF16(A, Bf, C) \
  __builtin_amdgcn_wmma_f32_16x16x32_bf16(false, (A), false, (Bf), (short)0, (C), false, false)

// ---------------------------------------------------------------------------
// f32 -> bf16 conversion
// ---------------------------------------------------------------------------
__global__ void cvt_f32_bf16(const float* __restrict__ src, __bf16* __restrict__ dst, int n) {
  int i = blockIdx.x * blockDim.x + threadIdx.x;
  int stride = gridDim.x * blockDim.x;
  for (; i < n; i += stride) dst[i] = f2bf(src[i]);
}

// ---------------------------------------------------------------------------
// QKV projection: [8192,512] x [512,1536] + bias -> Q/K/V bf16 [B,H,S,D]
// 128x128 block tile, 8 waves (2x4), wave tile 64x32, K-step 32.
// A-tile staged by TDM (contiguous 2D tile); B-tile transposed manually.
// ---------------------------------------------------------------------------
__global__ __launch_bounds__(256) void qkv_gemm(
    const __bf16* __restrict__ xb,    // [8192,512]
    const __bf16* __restrict__ wb,    // [512,1536]
    const float*  __restrict__ bias,  // [1536]
    __bf16* __restrict__ Qb, __bf16* __restrict__ Kb, __bf16* __restrict__ Vb) {
  const int K = EE, N = 3 * EE;
  __shared__ __bf16 sA[128 * 32];
  __shared__ __bf16 sB[128 * 32];  // transposed [n][k]

  int tid = threadIdx.x, lane = tid & 31, wave = tid >> 5;
  int wm = wave >> 2, wn = wave & 3;
  int m0 = blockIdx.y * 128, n0 = blockIdx.x * 128;

  v8f zero = {};
  v8f acc[4][2];
#pragma unroll
  for (int mf = 0; mf < 4; ++mf)
#pragma unroll
    for (int nf = 0; nf < 2; ++nf) acc[mf][nf] = zero;

  for (int k0 = 0; k0 < K; k0 += 32) {
    __syncthreads();
#if HAS_TDM
    if (wave == 0)  // DMA A tile: 128 rows x 32 k, row stride 512
      tdm_load_2d(lds_off_of(sA), xb + (size_t)m0 * K + k0, K, 8192, 32, 128, K);
#else
    {  // stage A (row-major): 128 rows x 32 k, 16 bf16 per thread
      int r = tid >> 1, c = (tid & 1) * 16;
      const uint4* src = (const uint4*)(xb + (size_t)(m0 + r) * K + k0 + c);
      uint4* dst = (uint4*)(sA + r * 32 + c);
      dst[0] = src[0]; dst[1] = src[1];
    }
#endif
    {  // stage B transposed: sB[n][k] = wb[k0+k][n0+n]
      int k = tid & 31, nb = (tid >> 5) * 16;
      const __bf16* src = wb + (size_t)(k0 + k) * N + n0 + nb;
#pragma unroll
      for (int j = 0; j < 16; ++j) sB[(nb + j) * 32 + k] = src[j];
      if (k0 + 32 < K)  // global_prefetch_b8 of next K-step weight rows
        __builtin_prefetch(wb + (size_t)(k0 + 32 + k) * N + n0 + nb, 0, 0);
    }
    WAIT_TENSOR();
    __syncthreads();

    v16bf af[4], bfr[2];
#pragma unroll
    for (int mf = 0; mf < 4; ++mf) af[mf] = load_frag_a(sA + (wm * 64 + mf * 16) * 32, 32, lane);
#pragma unroll
    for (int nf = 0; nf < 2; ++nf) bfr[nf] = load_frag_b(sB + (wn * 32 + nf * 16) * 32, 32, lane);
#pragma unroll
    for (int mf = 0; mf < 4; ++mf)
#pragma unroll
      for (int nf = 0; nf < 2; ++nf)
        acc[mf][nf] = WMMA_BF16(af[mf], bfr[nf], acc[mf][nf]);
  }

  // epilogue: add bias, scatter bf16 into Q/K/V [B,H,S,D]
  int n16 = lane & 15, hi = (lane >> 4) & 1;
#pragma unroll
  for (int mf = 0; mf < 4; ++mf)
#pragma unroll
    for (int nf = 0; nf < 2; ++nf) {
      int gn = n0 + wn * 32 + nf * 16 + n16;
      float bv = bias[gn];
      int sel = gn / EE;
      int e = gn - sel * EE;
      int h = e >> 6, d = e & 63;
      __bf16* dst = (sel == 0) ? Qb : ((sel == 1) ? Kb : Vb);
#pragma unroll
      for (int i = 0; i < 8; ++i) {
        int gm = m0 + wm * 64 + mf * 16 + i + hi * 8;
        int b = gm >> 12, s = gm & (SS - 1);
        dst[(((size_t)(b * HH + h) * SS + s) << 6) + d] = f2bf(acc[mf][nf][i] + bv);
      }
    }
}

// ---------------------------------------------------------------------------
// Flash attention: one block = 128 q rows of one (b,h); 8 waves x 16 rows.
// Q and K tiles staged by TDM (K's [key][d] layout IS the B-frag layout for
// QK^T); V transposed manually. Online softmax, bf16 WMMA throughout.
// ---------------------------------------------------------------------------
__global__ __launch_bounds__(256) void attn_fwd(
    const __bf16* __restrict__ Qb, const __bf16* __restrict__ Kb,
    const __bf16* __restrict__ Vb, __bf16* __restrict__ ctx) {
  __shared__ __bf16 sQ[128 * 64];
  __shared__ __bf16 sK[64 * 64];     // [key][d]  -> B-frag for QK^T
  __shared__ __bf16 sVt[64 * 64];    // [d][key]  -> B-frag for PV
  __shared__ __bf16 sP[8 * 16 * 64]; // per-wave P tile (C-layout -> A-layout)

  int tid = threadIdx.x, lane = tid & 31, wave = tid >> 5;
  int n16 = lane & 15, hi = (lane >> 4) & 1;
  int qb = blockIdx.x, h = blockIdx.y, b = blockIdx.z;
  int bh = b * HH + h;
  const __bf16* Qg = Qb + ((size_t)bh * SS << 6);
  const __bf16* Kg = Kb + ((size_t)bh * SS << 6);
  const __bf16* Vg = Vb + ((size_t)bh * SS << 6);
  int q0 = qb * 128;

#if HAS_TDM
  if (wave == 0)  // DMA Q tile: 128 x 64 contiguous
    tdm_load_2d(lds_off_of(sQ), Qg + ((size_t)q0 << 6), 64, SS, 64, 128, 64);
  WAIT_TENSOR();
#else
  {  // stage Q tile: 128 x 64
    int r = tid >> 1, c = (tid & 1) * 32;
    const uint4* src = (const uint4*)(Qg + ((size_t)(q0 + r) << 6) + c);
    uint4* dst = (uint4*)(sQ + r * 64 + c);
    dst[0] = src[0]; dst[1] = src[1]; dst[2] = src[2]; dst[3] = src[3];
  }
#endif
  __syncthreads();
  v16bf aq0 = load_frag_a(sQ + (wave * 16) * 64, 64, lane);       // d 0..31
  v16bf aq1 = load_frag_a(sQ + (wave * 16) * 64 + 32, 64, lane);  // d 32..63

  v8f zero = {};
  v8f o[4];
  float mrow[8], lrow[8];
#pragma unroll
  for (int df = 0; df < 4; ++df) o[df] = zero;
#pragma unroll
  for (int i = 0; i < 8; ++i) { mrow[i] = -1e30f; lrow[i] = 0.f; }

  __bf16* pw = sP + wave * 16 * 64;
  const float scale = 0.125f;  // 1/sqrt(64)

  for (int kb = 0; kb < SS / 64; ++kb) {
    __syncthreads();
#if HAS_TDM
    if (wave == 0)  // DMA K block: 64 keys x 64 d, contiguous
      tdm_load_2d(lds_off_of(sK), Kg + ((size_t)(kb * 64) << 6), 64, SS, 64, 64, 64);
#else
    {  // stage K block [key][d]
      int r = tid >> 2, c = (tid & 3) * 16;
      const uint4* src = (const uint4*)(Kg + ((size_t)(kb * 64 + r) << 6) + c);
      uint4* dst = (uint4*)(sK + r * 64 + c);
      dst[0] = src[0]; dst[1] = src[1];
    }
#endif
    {  // stage V block transposed [d][key]
      int r = tid >> 2, c = (tid & 3) * 16;
      const __bf16* src = Vg + ((size_t)(kb * 64 + r) << 6) + c;
#pragma unroll
      for (int j = 0; j < 16; ++j) sVt[(c + j) * 64 + r] = src[j];
    }
    WAIT_TENSOR();
    __syncthreads();

    // scores: 16 q rows x 64 keys, 4 n-frags, 2 K-steps over d
    v8f sacc[4];
#pragma unroll
    for (int nf = 0; nf < 4; ++nf) {
      sacc[nf] = zero;
      v16bf b0 = load_frag_b(sK + (nf * 16) * 64, 64, lane);       // d 0..31
      v16bf b1 = load_frag_b(sK + (nf * 16) * 64 + 32, 64, lane);  // d 32..63
      sacc[nf] = WMMA_BF16(aq0, b0, sacc[nf]);
      sacc[nf] = WMMA_BF16(aq1, b1, sacc[nf]);
    }

    // online softmax (rows live across lanes sharing lane>>4)
#pragma unroll
    for (int i = 0; i < 8; ++i) {
      float mx = -1e30f;
#pragma unroll
      for (int nf = 0; nf < 4; ++nf) {
        sacc[nf][i] *= scale;
        mx = fmaxf(mx, sacc[nf][i]);
      }
#pragma unroll
      for (int off = 1; off < 16; off <<= 1) mx = fmaxf(mx, __shfl_xor(mx, off, 32));
      float mnew = fmaxf(mrow[i], mx);
      float corr = __expf(mrow[i] - mnew);
      mrow[i] = mnew;
      float sum = 0.f;
#pragma unroll
      for (int nf = 0; nf < 4; ++nf) {
        float p = __expf(sacc[nf][i] - mnew);
        sacc[nf][i] = p;
        sum += p;
      }
#pragma unroll
      for (int off = 1; off < 16; off <<= 1) sum += __shfl_xor(sum, off, 32);
      lrow[i] = lrow[i] * corr + sum;
#pragma unroll
      for (int df = 0; df < 4; ++df) o[df][i] *= corr;
    }

    // C-layout f32 -> A-layout bf16 via per-wave LDS tile
#pragma unroll
    for (int nf = 0; nf < 4; ++nf)
#pragma unroll
      for (int i = 0; i < 8; ++i)
        pw[(i + hi * 8) * 64 + nf * 16 + n16] = f2bf(sacc[nf][i]);
    asm volatile("s_wait_dscnt 0" ::: "memory");

    v16bf p0 = load_frag_a(pw, 64, lane);       // keys 0..31
    v16bf p1 = load_frag_a(pw + 32, 64, lane);  // keys 32..63
#pragma unroll
    for (int df = 0; df < 4; ++df) {
      v16bf v0 = load_frag_b(sVt + (df * 16) * 64, 64, lane);
      v16bf v1 = load_frag_b(sVt + (df * 16) * 64 + 32, 64, lane);
      o[df] = WMMA_BF16(p0, v0, o[df]);
      o[df] = WMMA_BF16(p1, v1, o[df]);
    }
  }

  // normalize + write context bf16 as [B,S,E] (E = h*64 + d)
#pragma unroll
  for (int df = 0; df < 4; ++df)
#pragma unroll
    for (int i = 0; i < 8; ++i) {
      int qr = q0 + wave * 16 + i + hi * 8;
      ctx[((size_t)(b * SS + qr) << 9) + h * 64 + df * 16 + n16] = f2bf(o[df][i] / lrow[i]);
    }
}

// ---------------------------------------------------------------------------
// Output projection: ctx [8192,512] x w_out [512,512] + b_out -> f32 out
// ---------------------------------------------------------------------------
__global__ __launch_bounds__(256) void out_gemm(
    const __bf16* __restrict__ cb,   // [8192,512]
    const __bf16* __restrict__ wb,   // [512,512]
    const float*  __restrict__ bias, // [512]
    float* __restrict__ out) {
  const int K = EE, N = EE;
  __shared__ __bf16 sA[128 * 32];
  __shared__ __bf16 sB[128 * 32];

  int tid = threadIdx.x, lane = tid & 31, wave = tid >> 5;
  int wm = wave >> 2, wn = wave & 3;
  int m0 = blockIdx.y * 128, n0 = blockIdx.x * 128;

  v8f zero = {};
  v8f acc[4][2];
#pragma unroll
  for (int mf = 0; mf < 4; ++mf)
#pragma unroll
    for (int nf = 0; nf < 2; ++nf) acc[mf][nf] = zero;

  for (int k0 = 0; k0 < K; k0 += 32) {
    __syncthreads();
#if HAS_TDM
    if (wave == 0)
      tdm_load_2d(lds_off_of(sA), cb + (size_t)m0 * K + k0, K, 8192, 32, 128, K);
#else
    {
      int r = tid >> 1, c = (tid & 1) * 16;
      const uint4* src = (const uint4*)(cb + (size_t)(m0 + r) * K + k0 + c);
      uint4* dst = (uint4*)(sA + r * 32 + c);
      dst[0] = src[0]; dst[1] = src[1];
    }
#endif
    {
      int k = tid & 31, nb = (tid >> 5) * 16;
      const __bf16* src = wb + (size_t)(k0 + k) * N + n0 + nb;
#pragma unroll
      for (int j = 0; j < 16; ++j) sB[(nb + j) * 32 + k] = src[j];
      if (k0 + 32 < K)
        __builtin_prefetch(wb + (size_t)(k0 + 32 + k) * N + n0 + nb, 0, 0);
    }
    WAIT_TENSOR();
    __syncthreads();

    v16bf af[4], bfr[2];
#pragma unroll
    for (int mf = 0; mf < 4; ++mf) af[mf] = load_frag_a(sA + (wm * 64 + mf * 16) * 32, 32, lane);
#pragma unroll
    for (int nf = 0; nf < 2; ++nf) bfr[nf] = load_frag_b(sB + (wn * 32 + nf * 16) * 32, 32, lane);
#pragma unroll
    for (int mf = 0; mf < 4; ++mf)
#pragma unroll
      for (int nf = 0; nf < 2; ++nf)
        acc[mf][nf] = WMMA_BF16(af[mf], bfr[nf], acc[mf][nf]);
  }

  int n16 = lane & 15, hi = (lane >> 4) & 1;
#pragma unroll
  for (int mf = 0; mf < 4; ++mf)
#pragma unroll
    for (int nf = 0; nf < 2; ++nf) {
      int gn = n0 + wn * 32 + nf * 16 + n16;
      float bv = bias[gn];
#pragma unroll
      for (int i = 0; i < 8; ++i) {
        int gm = m0 + wm * 64 + mf * 16 + i + hi * 8;
        out[(size_t)gm * N + gn] = acc[mf][nf][i] + bv;
      }
    }
}

// ---------------------------------------------------------------------------
extern "C" void kernel_launch(void* const* d_in, const int* in_sizes, int n_in,
                              void* d_out, int out_size, void* d_ws, size_t ws_size,
                              hipStream_t stream) {
  const float* x     = (const float*)d_in[0];
  const float* w_qkv = (const float*)d_in[1];
  const float* b_qkv = (const float*)d_in[2];
  const float* w_out = (const float*)d_in[3];
  const float* b_out = (const float*)d_in[4];
  float* out = (float*)d_out;
  (void)in_sizes; (void)n_in; (void)out_size; (void)ws_size;

  char* ws = (char*)d_ws;
  size_t off = 0;
  auto alloc = [&](size_t elems) -> __bf16* {
    __bf16* p = (__bf16*)(ws + off);
    off += (elems * sizeof(__bf16) + 255) & ~(size_t)255;
    return p;
  };
  const size_t NX = (size_t)BB * SS * EE;       // 4,194,304
  __bf16* xb    = alloc(NX);
  __bf16* wqkvb = alloc((size_t)EE * 3 * EE);   // 786,432
  __bf16* woutb = alloc((size_t)EE * EE);       // 262,144
  __bf16* Qb    = alloc(NX);
  __bf16* Kb    = alloc(NX);
  __bf16* Vb    = alloc(NX);
  __bf16* ctx   = alloc(NX);

  cvt_f32_bf16<<<2048, 256, 0, stream>>>(x, xb, (int)NX);
  cvt_f32_bf16<<<768, 256, 0, stream>>>(w_qkv, wqkvb, EE * 3 * EE);
  cvt_f32_bf16<<<256, 256, 0, stream>>>(w_out, woutb, EE * EE);

  qkv_gemm<<<dim3(12, 64), 256, 0, stream>>>(xb, wqkvb, b_qkv, Qb, Kb, Vb);
  attn_fwd<<<dim3(SS / 128, HH, BB), 256, 0, stream>>>(Qb, Kb, Vb, ctx);
  out_gemm<<<dim3(4, 64), 256, 0, stream>>>(ctx, woutb, b_out, out);
}
